// BinaryTreeLSTMCell_34437047779344
// MI455X (gfx1250) — matrix operand order repaired and checked by reading, not compiled, not run on hardware
//
#include <hip/hip_runtime.h>

// ---------------------------------------------------------------------------
// Binary TreeLSTM, level-synchronous, WMMA f32_16x16x32_f16 gate GEMMs.
// B=128, L=256, H=512, DEPTH=8.  Gate GEMM per node: z(2560) = [Wl|Wr](2560x1024) @ [hl;hr] + b
// ---------------------------------------------------------------------------

typedef _Float16 f16;
typedef __attribute__((ext_vector_type(8)))  _Float16 v8h;
typedef __attribute__((ext_vector_type(16))) _Float16 v16h;
typedef __attribute__((ext_vector_type(8)))  float    v8f;
typedef __attribute__((ext_vector_type(4)))  float    v4f;

#define BATCH   128
#define SEQL    256
#define HDIM    512
#define G5      2560          // 5 gates * H
#define K2      1024          // hl(512) + hr(512)
#define OUT_ELEMS (BATCH * SEQL * HDIM)      // 16,777,216
#define FH_OFF    OUT_ELEMS                   // forest_h
#define FC_OFF    (OUT_ELEMS + BATCH * HDIM)  // forest_c

__device__ __forceinline__ float sigf(float x) { return 1.0f / (1.0f + __expf(-x)); }

__device__ __forceinline__ v8f wmma16x16x32(v16h a, v16h b, v8f c) {
  // (neg_a, A, neg_b, B, c_mod, C, reuse_a, reuse_b)
  return __builtin_amdgcn_wmma_f32_16x16x32_f16(false, a, false, b, (short)0, c,
                                                false, false);
}

// ---------------------------------------------------------------------------
// Prep: Wc[g][k] f16 row-major, g in [0,2560) = gate*512+row, k in [0,1024):
//   k<512 -> U_{gate}_l[row][k], else U_{gate}_r[row][k-512].  Also bias concat.
// ---------------------------------------------------------------------------
struct GatePtrs {
  const float* l[5];
  const float* r[5];
  const float* b[5];
};

__global__ __launch_bounds__(256) void prep_weights(GatePtrs gp, f16* __restrict__ Wc,
                                                    float* __restrict__ bias) {
  int idx = blockIdx.x * 256 + threadIdx.x;          // one element each
  if (idx < G5 * K2) {
    int g = idx >> 10;          // weight row 0..2559
    int k = idx & (K2 - 1);
    int gate = g >> 9;          // 0..4
    int row  = g & (HDIM - 1);
    float v = (k < HDIM) ? gp.l[gate][row * HDIM + k]
                         : gp.r[gate][row * HDIM + (k - HDIM)];
    Wc[idx] = (f16)v;
  }
  if (idx < G5) {
    int gate = idx >> 9;
    bias[idx] = gp.b[gate][idx & (HDIM - 1)];
  }
}

// ---------------------------------------------------------------------------
// Leaves: h0[b][l][f] = f16(emb[word_idx[b][l]][f]); 8 elements / thread.
// ---------------------------------------------------------------------------
__global__ __launch_bounds__(256) void leaf_gather(const int* __restrict__ word_idx,
                                                   const float* __restrict__ emb,
                                                   f16* __restrict__ h0) {
  long t = (long)blockIdx.x * 256 + threadIdx.x;     // per 8 elements
  long e = t * 8;                                    // < 16,777,216
  int tok = (int)(e >> 9);                           // b*L + l
  int f   = (int)(e & (HDIM - 1));
  int w = word_idx[tok];
  const float* src = emb + (long)w * HDIM + f;
  v4f a = *(const v4f*)(src);
  v4f b = *(const v4f*)(src + 4);
  v8h h;
#pragma unroll
  for (int i = 0; i < 4; ++i) { h[i] = (f16)a[i]; h[4 + i] = (f16)b[i]; }
  *(v8h*)(h0 + e) = h;
}

// ---------------------------------------------------------------------------
// One tree level.  n = nodes/tree at this level; M = B*n instance rows.
// Wave tile: 16 rows x 16 features, 5 gate accumulators (one A frag shared
// by 5 WMMAs per K-step).  Fully fused LSTM epilogue + post-order scatter.
//
// Fragment layouts used by this code (g = lane>>4, lm = lane&15):
//   A (16x32 f16): lane row M=lm; half  hw    -> K = kk+16g+hw   (contiguous 16)
//   B (32x16 f16): lane col N=lm; half  hw    -> K = kk+16g+hw   (contiguous 16)
//   D (16x16 f32): vgpr r -> M = r+8g, N = lm
// Post-order slot of internal node (level k, idx j):
//   pos = (j+1)*2^k - 2 - popcount(j);  out[:,1+pos]; node (1,0) also -> out[:,0]
// ---------------------------------------------------------------------------
template <int LVL>
__global__ __launch_bounds__(256) void tree_level(const f16*  __restrict__ h_prev,
                                                  const float* __restrict__ c_prev,
                                                  const f16*  __restrict__ Wc,
                                                  const float* __restrict__ bias,
                                                  f16*  __restrict__ h_cur,
                                                  float* __restrict__ c_cur,
                                                  float* __restrict__ out) {
  constexpr int n  = SEQL >> LVL;     // nodes per tree this level
  constexpr int ln = 8 - LVL;         // log2(n)

  const int lane = threadIdx.x & 31;
  const int wave = threadIdx.x >> 5;
  const int fBase = (blockIdx.x * 4 + (wave & 3)) * 16;   // feature tile (0..511)
  const int mBase = (blockIdx.y * 2 + (wave >> 2)) * 16;  // row tile
  const int g  = lane >> 4;
  const int lm = lane & 15;

  // --- A row pointer: X[m] = h_prev[b][2j .. 2j+1][:] (contiguous 1024 f16) ---
  const int m0 = mBase + lm;
  const int b0 = m0 >> ln;
  const int j0 = m0 & (n - 1);
  const f16* xrow = h_prev + (size_t)(b0 * (n << 1) + (j0 << 1)) * HDIM;
  const f16* xg0 = xrow + 16 * g;
  const f16* xg1 = xrow + 16 * g + 8;

  // --- B row pointers, one weight row per gate ---
  const f16* wr[5];
#pragma unroll
  for (int t = 0; t < 5; ++t)
    wr[t] = Wc + (size_t)(t * HDIM + fBase + lm) * K2 + 16 * g;

  v8f acc[5] = {};

#pragma unroll 4
  for (int kk = 0; kk < K2; kk += 32) {
    v8h a0 = *(const v8h*)(xg0 + kk);
    v8h a1 = *(const v8h*)(xg1 + kk);
    v16h a;
#pragma unroll
    for (int i = 0; i < 8; ++i) { a[i] = a0[i]; a[8 + i] = a1[i]; }
#pragma unroll
    for (int t = 0; t < 5; ++t) {
      v16h bf = *(const v16h*)(wr[t] + kk);
      acc[t] = wmma16x16x32(a, bf, acc[t]);
    }
  }

  // --- fused LSTM epilogue ---
  const int f = fBase + lm;
  const float bi  = bias[f];
  const float bfl = bias[HDIM + f];
  const float bfr = bias[2 * HDIM + f];
  const float bo  = bias[3 * HDIM + f];
  const float bc  = bias[4 * HDIM + f];

#pragma unroll
  for (int r = 0; r < 8; ++r) {
    const int mm = mBase + r + 8 * g;
    const int bb = mm >> ln;
    const int jj = mm & (n - 1);

    float zi  = acc[0][r] + bi;
    float zfl = acc[1][r] + bfl;
    float zfr = acc[2][r] + bfr;
    float zo  = acc[3][r] + bo;
    float zc  = acc[4][r] + bc;

    float cl, cr;
    if constexpr (LVL == 1) {
      cl = 0.5f; cr = 0.5f;                      // leaf c = 0.5
    } else {
      size_t cb = (size_t)(bb * (n << 1) + (jj << 1)) * HDIM + f;
      cl = c_prev[cb];
      cr = c_prev[cb + HDIM];
    }

    float cph = sigf(zi) * tanhf(zc) + sigf(zfl) * cl + sigf(zfr) * cr;
    float hph = sigf(zo) * tanhf(cph);

    if constexpr (LVL < 8) {
      h_cur[(size_t)(bb * n + jj) * HDIM + f] = (f16)hph;
      c_cur[(size_t)(bb * n + jj) * HDIM + f] = cph;
    }

    const int pos = ((jj + 1) << LVL) - 2 - __popc(jj);
    out[(size_t)(bb * SEQL + 1 + pos) * HDIM + f] = hph;
    if (LVL == 1 && jj == 0)
      out[(size_t)(bb * SEQL) * HDIM + f] = hph;   // torch-bug duplicate slot 0
    if constexpr (LVL == 8) {
      out[FH_OFF + (size_t)bb * HDIM + f] = hph;   // forest_h
      out[FC_OFF + (size_t)bb * HDIM + f] = cph;   // forest_c
    }
  }
}

// ---------------------------------------------------------------------------
// Workspace layout (bytes, all 256-aligned):
//   Wc   f16  2560*1024*2 =  5,242,880
//   bias f32  2560*4      =     10,240
//   h0   f16  128*256*512*2 = 33,554,432   (leaves)
//   hA   f16  128*128*512*2 = 16,777,216   (ping)
//   hB   f16  128* 64*512*2 =  8,388,608   (pong)
//   cA   f32  128*128*512*4 = 33,554,432
//   cB   f32  128* 64*512*4 = 16,777,216
// total ~114.3 MB
// ---------------------------------------------------------------------------
extern "C" void kernel_launch(void* const* d_in, const int* in_sizes, int n_in,
                              void* d_out, int out_size, void* d_ws, size_t ws_size,
                              hipStream_t stream) {
  (void)in_sizes; (void)n_in; (void)out_size; (void)ws_size;

  const int*   word_idx = (const int*)d_in[0];
  const float* emb      = (const float*)d_in[1];

  GatePtrs gp;
  // l-side: U_i_l, U_fl_l, U_fr_l, U_o_l, U_c_l  (d_in 2,4,6,8,10)
  // r-side: U_i_r, U_fl_r, U_fr_r, U_o_r, U_c_r  (d_in 3,5,7,9,11)
  // biases: b_i, b_fl, b_fr, b_o, b_c            (d_in 12..16)
  for (int t = 0; t < 5; ++t) {
    gp.l[t] = (const float*)d_in[2 + 2 * t];
    gp.r[t] = (const float*)d_in[3 + 2 * t];
    gp.b[t] = (const float*)d_in[12 + t];
  }

  char* ws = (char*)d_ws;
  f16*   Wc   = (f16*)(ws + 0);
  float* bias = (float*)(ws + 5242880);
  f16*   h0   = (f16*)(ws + 5253120);
  f16*   hA   = (f16*)(ws + 38807552);
  f16*   hB   = (f16*)(ws + 55584768);
  float* cA   = (float*)(ws + 63973376);
  float* cB   = (float*)(ws + 97527808);

  float* out = (float*)d_out;

  prep_weights<<<(G5 * K2) / 256, 256, 0, stream>>>(gp, Wc, bias);
  leaf_gather<<<(BATCH * SEQL * HDIM / 8) / 256, 256, 0, stream>>>(word_idx, emb, h0);

  // grid: x = 512/16/4 = 8 feature-blocks, y = (B*n)/16/2 row-blocks
  tree_level<1><<<dim3(8, 4 * 128), 256, 0, stream>>>(h0, nullptr, Wc, bias, hA, cA, out);
  tree_level<2><<<dim3(8, 4 *  64), 256, 0, stream>>>(hA, cA, Wc, bias, hB, cB, out);
  tree_level<3><<<dim3(8, 4 *  32), 256, 0, stream>>>(hB, cB, Wc, bias, hA, cA, out);
  tree_level<4><<<dim3(8, 4 *  16), 256, 0, stream>>>(hA, cA, Wc, bias, hB, cB, out);
  tree_level<5><<<dim3(8, 4 *   8), 256, 0, stream>>>(hB, cB, Wc, bias, hA, cA, out);
  tree_level<6><<<dim3(8, 4 *   4), 256, 0, stream>>>(hA, cA, Wc, bias, hB, cB, out);
  tree_level<7><<<dim3(8, 4 *   2), 256, 0, stream>>>(hB, cB, Wc, bias, hA, cA, out);
  tree_level<8><<<dim3(8, 4 *   1), 256, 0, stream>>>(hA, cA, Wc, bias, hB, cB, out);
}
